// CustomLSTM_8778913153263
// MI455X (gfx1250) — compile-verified, matrix operand order checked by reading, not run on hardware
//
#include <hip/hip_runtime.h>
#include <hip/hip_bf16.h>
#include <math.h>

// ---------------------------------------------------------------------------
// LSTM (B=128, S=1000, I=128, H=1024) + linear head (1024 -> 50)
// Strategy:
//   * fuse [x_t | h] @ [Wx ; Wh] into one K=1152 GEMM per timestep
//   * f16 WMMA (v_wmma_f32_16x16x32_f16), f32 accumulation
//   * persistent kernel, 64 WGs x 256 threads; ONE device barrier per step
//   * each WG's 147KB weight block is fully resident in LDS (320KB/WGP),
//     loaded once via global_load_async_to_lds_b128 (ASYNCcnt path)
//   * c-state lives in registers; h ping-pongs through L2 in f16
// ---------------------------------------------------------------------------

typedef __attribute__((ext_vector_type(16))) _Float16 v16h;
typedef __attribute__((ext_vector_type(8)))  _Float16 v8h;
typedef __attribute__((ext_vector_type(8)))  float    v8f;

#define BATCH   128
#define INPUT   128
#define HID     1024
#define KDIM    (INPUT + HID)          // 1152
#define NGATES  4
#define GCOLS   (NGATES * HID)         // 4096
#define CATS    50
#define NWG     64                     // workgroups in persistent kernel
#define KT      (KDIM / 32)            // 36 k-tiles of 32

// workspace layout (bytes)
#define OFF_WT   ((size_t)0)                                   // 4096*1152 f16
#define OFF_BIAS (OFF_WT   + (size_t)GCOLS * KDIM * 2)         // 4096 f32
#define OFF_X16  (OFF_BIAS + (size_t)GCOLS * 4)                // S*128*128 f16
#define OFF_H16(S) (OFF_X16 + (size_t)(S) * BATCH * INPUT * 2) // 2*128*1024 f16
#define OFF_BAR(S) (OFF_H16(S) + (size_t)2 * BATCH * HID * 2)  // ints
#define HBUF ((size_t)BATCH * HID)

// ---- fragment helper (16-bit A/B 16x32 VGPR layout, ISA 7.12.2) -----------
// lane L: row/col = L&15, holds K = koff..koff+7 (elems 0..7) and
//                        K = koff+16..koff+23 (elems 8..15), koff=(L>>4)*8
__device__ inline v16h frag_ld_16x32(const _Float16* base, int ld, int lane) {
    int r    = lane & 15;
    int koff = (lane >> 4) * 8;
    const _Float16* p = base + (size_t)r * ld + koff;
    v8h lo = *(const v8h*)(p);
    v8h hi = *(const v8h*)(p + 16);
    v16h f;
#pragma unroll
    for (int i = 0; i < 8; ++i) { f[i] = lo[i]; f[i + 8] = hi[i]; }
    return f;
}

__device__ inline float sigmoidf(float x) { return 1.0f / (1.0f + __expf(-x)); }

// ---- device-wide barrier (sense/generation) -------------------------------
__device__ inline void grid_barrier(int* cnt, int* gen, int nwg) {
    __builtin_amdgcn_fence(__ATOMIC_RELEASE, "agent");   // flush h stores
    __syncthreads();
    if (threadIdx.x == 0) {
        int g = __hip_atomic_load(gen, __ATOMIC_RELAXED, __HIP_MEMORY_SCOPE_AGENT);
        int arrived = __hip_atomic_fetch_add(cnt, 1, __ATOMIC_ACQ_REL,
                                             __HIP_MEMORY_SCOPE_AGENT) + 1;
        if (arrived == nwg) {
            __hip_atomic_store(cnt, 0, __ATOMIC_RELAXED, __HIP_MEMORY_SCOPE_AGENT);
            __hip_atomic_store(gen, g + 1, __ATOMIC_RELEASE, __HIP_MEMORY_SCOPE_AGENT);
        } else {
            while (__hip_atomic_load(gen, __ATOMIC_ACQUIRE,
                                     __HIP_MEMORY_SCOPE_AGENT) == g) {
                __builtin_amdgcn_s_sleep(1);
            }
        }
    }
    __syncthreads();
    __builtin_amdgcn_fence(__ATOMIC_ACQUIRE, "agent");   // invalidate for new h
}

// ---- prep: fused transposed weights Wt[4096][1152] (f16), bias, h0, bar ---
__global__ __launch_bounds__(256) void prep_w_kernel(
    const float* Wii, const float* Whi, const float* bii,
    const float* Wif, const float* Whf, const float* bif,
    const float* Wig, const float* Whg, const float* big,
    const float* Wio, const float* Who, const float* bio,
    _Float16* Wt, float* bias4, _Float16* H16, int* bar)
{
    int idx = blockIdx.x * blockDim.x + threadIdx.x;
    const int TOT = GCOLS * KDIM;
    if (idx < TOT) {
        int j = idx / KDIM, k = idx % KDIM;       // j = gate-col, k = fused K
        int gate = j >> 10, u = j & 1023;
        const float* Wi[NGATES] = {Wii, Wif, Wig, Wio};
        const float* Wh[NGATES] = {Whi, Whf, Whg, Who};
        float v = (k < INPUT) ? Wi[gate][(size_t)k * HID + u]
                              : Wh[gate][(size_t)(k - INPUT) * HID + u];
        Wt[idx] = (_Float16)v;
    }
    if (idx < GCOLS) {
        int gate = idx >> 10, u = idx & 1023;
        const float* bb[NGATES] = {bii, bif, big, bio};
        bias4[idx] = bb[gate][u];
    }
    if (idx < 2 * BATCH * HID) H16[idx] = (_Float16)0.f;  // h0 = 0 (both bufs)
    if (idx < 8) bar[idx] = 0;
}

// ---- prep: x [B][S][I] f32 -> X16 [S][B][I] f16 ---------------------------
__global__ __launch_bounds__(256) void prep_x_kernel(const float* x,
                                                     _Float16* X16, int S)
{
    int idx = blockIdx.x * blockDim.x + threadIdx.x;
    int tot = BATCH * S * INPUT;
    if (idx >= tot) return;
    int k = idx & (INPUT - 1);
    int r = idx >> 7;             // s*B + b
    int b = r & (BATCH - 1);
    int s = r >> 7;
    X16[idx] = (_Float16)x[((size_t)b * S + s) * INPUT + k];
}

// ---- persistent LSTM kernel ----------------------------------------------
__global__ __launch_bounds__(256) void lstm_kernel(
    const _Float16* __restrict__ Wt, const float* __restrict__ bias4,
    const _Float16* __restrict__ X16, _Float16* H16, int* bar, int S)
{
    // entire per-WG weight block resident in LDS: 64 cols x 1152 K x f16
    __shared__ _Float16 sW[64 * KDIM];           // 147,456 bytes (< 320KB/WGP)

    const int t    = threadIdx.x;
    const int lane = t & 31;
    const int wave = t >> 5;              // 8 waves
    const int wg   = blockIdx.x;          // 64 WGs
    const int n_base = wg * 16;           // hidden units owned by this WG
    const int m_base = wave * 16;         // batch rows owned by this wave

    // ---- one-time: async-copy weight block global -> LDS (ASYNCcnt) ------
    {
        const int crow  = t >> 2;                 // 0..63  (gate*16 + col)
        const int cgate = crow >> 4, ccol = crow & 15;
        const _Float16* gsrc =
            Wt + (size_t)(cgate * HID + n_base + ccol) * KDIM;
        unsigned ldsrow = (unsigned)(uintptr_t)(&sW[crow * KDIM]);
#pragma unroll 4
        for (int j = 0; j < KDIM / 8 / 4; ++j) {  // 36 chunks of 8 halfs
            int ch = ((t & 3) * (KDIM / 8 / 4) + j) * 8;
            unsigned loff = ldsrow + (unsigned)(ch * 2);
            unsigned long long gaddr = (unsigned long long)(uintptr_t)(gsrc + ch);
            asm volatile("global_load_async_to_lds_b128 %0, %1, off"
                         :: "v"(loff), "v"(gaddr) : "memory");
        }
        asm volatile("s_wait_asynccnt 0x0" ::: "memory");
        __syncthreads();
    }

    // per-lane gate biases (column n, same across all 8 acc VGPRs)
    const int n  = n_base + (lane & 15);
    const float bi = bias4[0 * HID + n];
    const float bf = bias4[1 * HID + n];
    const float bg = bias4[2 * HID + n];
    const float bo = bias4[3 * HID + n];
    const int bsel = (lane >> 4) * 8;     // C-layout row offset for this lane

    v8f c_state = {};                     // cell state lives in registers

    for (int s = 0; s < S; ++s) {
        const _Float16* Xs     = X16 + (size_t)s * BATCH * INPUT;
        const _Float16* Hread  = H16 + (size_t)(s & 1) * HBUF;
        _Float16*       Hwrite = H16 + (size_t)((s & 1) ^ 1) * HBUF;

        v8f acc_i = {}, acc_f = {}, acc_g = {}, acc_o = {};

#pragma unroll
        for (int kt = 0; kt < KT; ++kt) {
            const int k0 = kt * 32;

            // A fragment: [x_t | h] row-major (global / L2)
            const _Float16* abase;
            int ld;
            if (k0 < INPUT) { abase = Xs + (size_t)m_base * INPUT + k0; ld = INPUT; }
            else            { abase = Hread + (size_t)m_base * HID + (k0 - INPUT); ld = HID; }
            v16h a = frag_ld_16x32(abase, ld, lane);

            // B fragments for the 4 gates straight from resident LDS block
            v16h b0 = frag_ld_16x32(&sW[(0 * 16) * KDIM + k0], KDIM, lane);
            v16h b1 = frag_ld_16x32(&sW[(1 * 16) * KDIM + k0], KDIM, lane);
            v16h b2 = frag_ld_16x32(&sW[(2 * 16) * KDIM + k0], KDIM, lane);
            v16h b3 = frag_ld_16x32(&sW[(3 * 16) * KDIM + k0], KDIM, lane);

            acc_i = __builtin_amdgcn_wmma_f32_16x16x32_f16(false, a, false, b0,
                        (short)0, acc_i, false, false);
            acc_f = __builtin_amdgcn_wmma_f32_16x16x32_f16(false, a, false, b1,
                        (short)0, acc_f, false, false);
            acc_g = __builtin_amdgcn_wmma_f32_16x16x32_f16(false, a, false, b2,
                        (short)0, acc_g, false, false);
            acc_o = __builtin_amdgcn_wmma_f32_16x16x32_f16(false, a, false, b3,
                        (short)0, acc_o, false, false);
        }

        // elementwise LSTM cell update; write h (f16) to the other buffer
#pragma unroll
        for (int v = 0; v < 8; ++v) {
            float gi = sigmoidf(acc_i[v] + bi);
            float gf = sigmoidf(acc_f[v] + bf);
            float gg = tanhf(acc_g[v] + bg);
            float go = sigmoidf(acc_o[v] + bo);
            float c  = gf * c_state[v] + gi * gg;
            c_state[v] = c;
            float h  = go * tanhf(c);
            int   b  = m_base + bsel + v;            // C-matrix VGPR layout
            Hwrite[(size_t)b * HID + n] = (_Float16)h;
        }

        grid_barrier(&bar[0], &bar[1], NWG);
    }
}

// ---- head: out[b][cat] = h_T[b] . W_out[cat] + b_out[cat] -----------------
__global__ __launch_bounds__(256) void head_kernel(
    const _Float16* __restrict__ Hfin, const float* __restrict__ W_out,
    const float* __restrict__ b_out, float* __restrict__ out)
{
    int idx = blockIdx.x * blockDim.x + threadIdx.x;
    if (idx >= BATCH * CATS) return;
    int b = idx / CATS, cat = idx % CATS;
    const _Float16* h = Hfin + (size_t)b * HID;
    const float*    w = W_out + (size_t)cat * HID;
    float acc = 0.f;
#pragma unroll 8
    for (int k = 0; k < HID; ++k) acc += (float)h[k] * w[k];
    out[idx] = acc + b_out[cat];
}

// ---------------------------------------------------------------------------
extern "C" void kernel_launch(void* const* d_in, const int* in_sizes, int n_in,
                              void* d_out, int out_size, void* d_ws, size_t ws_size,
                              hipStream_t stream) {
    const float* x    = (const float*)d_in[0];
    const float* Wii  = (const float*)d_in[1];
    const float* Whi  = (const float*)d_in[2];
    const float* bii  = (const float*)d_in[3];
    const float* Wif  = (const float*)d_in[4];
    const float* Whf  = (const float*)d_in[5];
    const float* bif  = (const float*)d_in[6];
    const float* Wig  = (const float*)d_in[7];
    const float* Whg  = (const float*)d_in[8];
    const float* big  = (const float*)d_in[9];
    const float* Wio  = (const float*)d_in[10];
    const float* Who  = (const float*)d_in[11];
    const float* bio  = (const float*)d_in[12];
    const float* Wout = (const float*)d_in[13];
    const float* bout = (const float*)d_in[14];

    const int S = in_sizes[0] / (BATCH * INPUT);   // 1000

    char* ws = (char*)d_ws;
    _Float16* Wt    = (_Float16*)(ws + OFF_WT);
    float*    bias4 = (float*)   (ws + OFF_BIAS);
    _Float16* X16   = (_Float16*)(ws + OFF_X16);
    _Float16* H16   = (_Float16*)(ws + OFF_H16(S));
    int*      bar   = (int*)     (ws + OFF_BAR(S));

    // 1) build fused transposed f16 weights + bias, zero h0 + barrier
    {
        int tot = GCOLS * KDIM;
        prep_w_kernel<<<(tot + 255) / 256, 256, 0, stream>>>(
            Wii, Whi, bii, Wif, Whf, bif, Wig, Whg, big, Wio, Who, bio,
            Wt, bias4, H16, bar);
    }
    // 2) x f32 [B][S][I] -> f16 [S][B][I]
    {
        int tot = BATCH * S * INPUT;
        prep_x_kernel<<<(tot + 255) / 256, 256, 0, stream>>>(x, X16, S);
    }
    // 3) persistent WMMA LSTM over S steps (one device barrier per step)
    lstm_kernel<<<NWG, 256, 0, stream>>>(Wt, bias4, X16, H16, bar, S);

    // 4) final h is in ping-pong buffer (S&1) reader-side: writes of step
    //    s go to ((s&1)^1); for even S the last write landed in buffer 0.
    const _Float16* Hfin = H16 + (size_t)(S & 1) * HBUF;
    head_kernel<<<(BATCH * CATS + 255) / 256, 256, 0, stream>>>(
        Hfin, Wout, bout, (float*)d_out);
}
